// GenericRNN_87342454932147
// MI455X (gfx1250) — matrix operand order repaired and checked by reading, not compiled, not run on hardware
//
#include <hip/hip_runtime.h>
#include <hip/hip_bf16.h>

// Problem constants
#define BB   32
#define TT   1024
#define DIN  1024
#define HH   512
#define G4   2048   // 4*H
#define LL   3
#define NWG  16     // workgroups per direction in the scan
#define SL   32     // h columns per scan workgroup (HH / NWG)

typedef __attribute__((ext_vector_type(16))) __bf16      v16bf;
typedef __attribute__((ext_vector_type(8)))  float       v8f;
typedef __attribute__((ext_vector_type(8)))  unsigned int v8u;

static __device__ __forceinline__ v8f wmma_bf16(v8u a, v8u b, v8f c) {
  return __builtin_amdgcn_wmma_f32_16x16x32_bf16(
      false, __builtin_bit_cast(v16bf, a),
      false, __builtin_bit_cast(v16bf, b),
      (short)0, c, false, false);
}

static __device__ __forceinline__ unsigned short f2bf(float f) {
  unsigned u = __builtin_bit_cast(unsigned, f);
  u += 0x7FFFu + ((u >> 16) & 1u);   // round-to-nearest-even
  return (unsigned short)(u >> 16);
}

static __device__ __forceinline__ float sigm(float x) {
  return 1.0f / (1.0f + __expf(-x));
}

// async copy of one 16B chunk: LDS[ldsb] = MEM[gsrc]  (ASYNCcnt tracked)
static __device__ __forceinline__ void async_b128(unsigned ldsb, const void* gsrc) {
  asm volatile("global_load_async_to_lds_b128 %0, %1, off"
               :: "v"(ldsb), "v"(gsrc) : "memory");
}
static __device__ __forceinline__ void wait_async0() {
  asm volatile("s_wait_asynccnt 0x0" ::: "memory");
}

// ---------------------------------------------------------------------------
// init: sequence lengths from paddings + zero the inter-WG barrier counters
// ---------------------------------------------------------------------------
__global__ void k_init(const float* __restrict__ pad,
                       int* __restrict__ lengths,
                       unsigned int* __restrict__ bar) {
  int t = threadIdx.x;
  if (t < 8) bar[t] = 0u;
  if (t < BB) {
    int n = 0;
    for (int i = 0; i < TT; ++i) n += (pad[t * TT + i] < 0.5f) ? 1 : 0;
    lengths[t] = n;
  }
}

// ---------------------------------------------------------------------------
// pack Wx [6][1024][2048] f32 -> bf16 WMMA B-fragment order:
//   [ld][ntile(128)][ktile(32)][lane(32)*16 + slot(16)]
// B order used here (contiguous k, as the A loads): VGPR v holds K=2v,2v+1 (lanes 0-15, N=lane),
// lanes 16-31 hold K=16+2v,17+2v (N=lane-16).
// ---------------------------------------------------------------------------
__global__ void k_pack_wx(const float* __restrict__ wx,
                          unsigned short* __restrict__ dst) {
  long i = (long)blockIdx.x * blockDim.x + threadIdx.x;
  const long TOT = 6L * 128 * 32 * 512;
  if (i >= TOT) return;
  int slot = (int)(i & 15);
  int lane = (int)((i >> 4) & 31);
  long rest = i >> 9;
  int kt = (int)(rest & 31); rest >>= 5;
  int ntile = (int)(rest & 127); rest >>= 7;
  int ld = (int)rest;
  int v = slot >> 1, p = slot & 1;
  int kl = (lane < 16) ? (2 * v + p) : (16 + 2 * v + p);
  int nl = lane & 15;
  long k = (long)kt * 32 + kl;
  long n = (long)ntile * 16 + nl;
  dst[i] = f2bf(wx[((long)ld * DIN + k) * G4 + n]);
}

// ---------------------------------------------------------------------------
// pack Wh [6][512][2048] f32 -> bf16 B-fragment order per scan slice:
//   [ld][slice(16)][gate(4)][nt2(2)][ktile(16)][512]
// gate-column n_global = gate*512 + slice*32 + nt2*16 + nl
// ---------------------------------------------------------------------------
__global__ void k_pack_wh(const float* __restrict__ wh,
                          unsigned short* __restrict__ dst) {
  long i = (long)blockIdx.x * blockDim.x + threadIdx.x;
  const long TOT = 6L * 16 * 4 * 2 * 16 * 512;
  if (i >= TOT) return;
  int slot = (int)(i & 15);
  int lane = (int)((i >> 4) & 31);
  long rest = i >> 9;
  int kt = (int)(rest & 15); rest >>= 4;
  int nt = (int)(rest & 1);  rest >>= 1;
  int g  = (int)(rest & 3);  rest >>= 2;
  int s  = (int)(rest & 15); rest >>= 4;
  int ld = (int)rest;
  int v = slot >> 1, p = slot & 1;
  int kl = (lane < 16) ? (2 * v + p) : (16 + 2 * v + p);
  int nl = lane & 15;
  long k = (long)kt * 32 + kl;
  long n = (long)g * HH + (long)s * SL + nt * 16 + nl;
  dst[i] = f2bf(wh[((long)ld * HH + k) * G4 + n]);
}

// fp32 -> bf16 activation convert
__global__ void k_cvt(const float* __restrict__ src,
                      unsigned short* __restrict__ dst, long n) {
  long i = (long)blockIdx.x * blockDim.x + threadIdx.x;
  if (i < n) dst[i] = f2bf(src[i]);
}

// ---------------------------------------------------------------------------
// Input projection GEMM: xg[M=32768, N=2048] = xb[M,1024] @ WxP + bias
// block = 256 threads (8 waves), block tile 128x128, wave tile 32x64
// ---------------------------------------------------------------------------
__global__ void __launch_bounds__(256) k_xproj(
    const unsigned short* __restrict__ xb,    // [32768,1024] bf16 row-major
    const unsigned short* __restrict__ wxp,   // packed B fragments
    const float* __restrict__ bias,           // [2048]
    float* __restrict__ xg)                   // [32768,2048] f32
{
  const int tid  = threadIdx.x;
  const int lane = tid & 31;
  const int w    = tid >> 5;
  const int wm   = w >> 1, wn = w & 1;
  const int hi   = lane >> 4;     // 0 or 1 (half-wave)
  const int ln   = lane & 15;
  const long mb  = (long)blockIdx.x * 128 + (long)wm * 32;
  const int nb16 = blockIdx.y * 8 + wn * 4;   // base N-tile index

  v8f acc[2][4];
#pragma unroll
  for (int a = 0; a < 2; ++a)
#pragma unroll
    for (int b = 0; b < 4; ++b)
#pragma unroll
      for (int r = 0; r < 8; ++r) acc[a][b][r] = 0.0f;

  for (int kt = 0; kt < 32; ++kt) {
    const int k0 = kt * 32;
    v8u afr[2];
#pragma unroll
    for (int mt = 0; mt < 2; ++mt) {
      long row = mb + mt * 16 + ln;
      const unsigned short* p = xb + row * DIN + k0 + hi * 16;
      uint4 lo = *(const uint4*)p;
      uint4 hj = *(const uint4*)(p + 8);
      v8u a;
      a[0] = lo.x; a[1] = lo.y; a[2] = lo.z; a[3] = lo.w;
      a[4] = hj.x; a[5] = hj.y; a[6] = hj.z; a[7] = hj.w;
      afr[mt] = a;
    }
#pragma unroll
    for (int nt = 0; nt < 4; ++nt) {
      long ntg = nb16 + nt;
      v8u b = *(const v8u*)(wxp + (ntg * 32 + kt) * 512 + lane * 16);
#pragma unroll
      for (int mt = 0; mt < 2; ++mt)
        acc[mt][nt] = wmma_bf16(afr[mt], b, acc[mt][nt]);
    }
  }

#pragma unroll
  for (int nt = 0; nt < 4; ++nt) {
    long n = (long)(nb16 + nt) * 16 + ln;
    float bv = bias[n];
#pragma unroll
    for (int mt = 0; mt < 2; ++mt)
#pragma unroll
      for (int r = 0; r < 8; ++r) {
        long m = mb + mt * 16 + r + hi * 8;
        xg[m * G4 + n] = acc[mt][nt][r] + bv;
      }
  }
}

// ---------------------------------------------------------------------------
// Recurrent scan, one direction. 16 workgroups (one per WGP), each owns
// 32 h-columns. Wh slice (128 KB) is async-preloaded into LDS once; per
// step the full h [32x512] bf16 is async-staged into LDS, gates-slice =
// h @ Wh-slice (WMMA from LDS) + xg, then the LSTM cell; h is exchanged
// through L2 with an agent-scope arrive/spin barrier.
// LDS: 128K (Wh) + 32K (h) + 16K (gates) ~ 180 KB -> one WG per WGP.
// ---------------------------------------------------------------------------
__global__ void __launch_bounds__(256) k_scan(
    const float* __restrict__ xg,             // [B*T, 2048]
    const unsigned short* __restrict__ whp,   // packed Wh for this (l,d)
    unsigned short* __restrict__ hbuf,        // [32,512] bf16 (this dir)
    unsigned short* __restrict__ xnext,       // [B*T, 1024] bf16
    float* __restrict__ dout,                 // [B*T, 1024] f32 or null
    const int* __restrict__ lengths,
    unsigned int* __restrict__ bar,           // one counter for this (l,d)
    int dir, int last)
{
  __shared__ __align__(32) unsigned short wh_lds[4 * 2 * 16 * 512]; // 128 KB
  __shared__ __align__(32) unsigned short h_lds[BB * HH];           // 32 KB
  __shared__ float gates_lds[4 * BB * SL];                          // 16 KB
  __shared__ int len_lds[BB];

  const int tid  = threadIdx.x;
  const int lane = tid & 31;
  const int w    = tid >> 5;
  const int g    = w >> 1;        // gate 0..3 (i,f,g,o)
  const int ntw  = w & 1;         // n-tile within slice
  const int hi   = lane >> 4, ln = lane & 15;
  const int s    = blockIdx.x;    // slice 0..15

  if (tid < BB) len_lds[tid] = lengths[tid];

  // async preload of this WG's Wh slice (131072 B, 512 B / thread)
  {
    const char* src = (const char*)(whp + (size_t)s * 65536) + tid * 512;
    unsigned dstb = (unsigned)(size_t)&wh_lds[0] + (unsigned)(tid * 512);
#pragma unroll
    for (int q = 0; q < 32; ++q)
      async_b128(dstb + q * 16, src + q * 16);
    wait_async0();
  }

  const unsigned short* wbase = wh_lds + (size_t)((g * 2 + ntw) * 16) * 512;
  const unsigned h_lds_base = (unsigned)(size_t)&h_lds[0];

  const int em   = tid >> 3;          // batch row 0..31 for elementwise
  const int ehc0 = (tid & 7) * 4;     // first of 4 h-columns
  float c[4] = {0.f, 0.f, 0.f, 0.f};
  __syncthreads();

  for (int t = 0; t < TT; ++t) {
    v8f acc[2];
#pragma unroll
    for (int mt = 0; mt < 2; ++mt)
#pragma unroll
      for (int r = 0; r < 8; ++r) acc[mt][r] = 0.f;

    if (t > 0) {
      // async-stage full h [32x512] bf16 from global into LDS (128 B/thread)
      const char* hsrc = (const char*)hbuf + tid * 128;
      unsigned hdst = h_lds_base + (unsigned)(tid * 128);
#pragma unroll
      for (int q = 0; q < 8; ++q)
        async_b128(hdst + q * 16, hsrc + q * 16);
      wait_async0();
      __syncthreads();

      for (int kt = 0; kt < 16; ++kt) {
        v8u bfr = *(const v8u*)(wbase + kt * 512 + lane * 16);
#pragma unroll
        for (int mt = 0; mt < 2; ++mt) {
          int row = mt * 16 + ln;
          const unsigned short* p = h_lds + row * HH + kt * 32 + hi * 16;
          uint4 lo = *(const uint4*)p;
          uint4 hj = *(const uint4*)(p + 8);
          v8u a;
          a[0] = lo.x; a[1] = lo.y; a[2] = lo.z; a[3] = lo.w;
          a[4] = hj.x; a[5] = hj.y; a[6] = hj.z; a[7] = hj.w;
          acc[mt] = wmma_bf16(a, bfr, acc[mt]);
        }
      }
      __syncthreads();
    }

    // write gate slice [32 x 32] for this gate to LDS
#pragma unroll
    for (int mt = 0; mt < 2; ++mt)
#pragma unroll
      for (int r = 0; r < 8; ++r) {
        int m = mt * 16 + r + hi * 8;
        gates_lds[(g * BB + m) * SL + ntw * 16 + ln] = acc[mt][r];
      }
    __syncthreads();

    // LSTM cell: each thread owns 4 (m, hc) cells
    int len = len_lds[em];
    int tt;
    if (dir == 0) tt = t;
    else { int vv = TT - 1 - t + len; tt = (vv >= TT) ? vv - TT : vv; }
    long rowbase = ((long)em * TT + tt) * G4 + (long)s * SL;
#pragma unroll
    for (int j = 0; j < 4; ++j) {
      int hc = ehc0 + j;
      float vi = gates_lds[(0 * BB + em) * SL + hc] + xg[rowbase + 0 * HH + hc];
      float vf = gates_lds[(1 * BB + em) * SL + hc] + xg[rowbase + 1 * HH + hc];
      float vg = gates_lds[(2 * BB + em) * SL + hc] + xg[rowbase + 2 * HH + hc];
      float vo = gates_lds[(3 * BB + em) * SL + hc] + xg[rowbase + 3 * HH + hc];
      float cn = sigm(vf) * c[j] + sigm(vi) * tanhf(vg);
      float hv = sigm(vo) * tanhf(cn);
      c[j] = cn;
      int hcol = s * SL + hc;
      hbuf[em * HH + hcol] = f2bf(hv);
      long oidx = ((long)em * TT + tt) * 1024 + (long)dir * HH + hcol;
      xnext[oidx] = f2bf(hv);
      if (last) dout[oidx] = hv;
    }

    // prefetch next step's xg row segments (read-once streaming data)
    if (t + 1 < TT) {
      int tn;
      if (dir == 0) tn = t + 1;
      else { int vv = TT - 2 - t + len; tn = (vv >= TT) ? vv - TT : vv; }
      const float* pf = xg + ((long)em * TT + tn) * G4 + (long)s * SL + ehc0;
#pragma unroll
      for (int gg = 0; gg < 4; ++gg)
        __builtin_prefetch(pf + gg * HH, 0, 1);
    }

    // inter-workgroup barrier: release h, arrive, spin, acquire
    __builtin_amdgcn_fence(__ATOMIC_RELEASE, "agent");
    __syncthreads();
    if (tid == 0) {
      __hip_atomic_fetch_add(bar, 1u, __ATOMIC_RELAXED, __HIP_MEMORY_SCOPE_AGENT);
      unsigned int tgt = (unsigned int)(NWG * (t + 1));
      while (__hip_atomic_load(bar, __ATOMIC_RELAXED, __HIP_MEMORY_SCOPE_AGENT) < tgt) {
        __builtin_amdgcn_s_sleep(1);
      }
    }
    __syncthreads();
    __builtin_amdgcn_fence(__ATOMIC_ACQUIRE, "agent");
  }
}

// ---------------------------------------------------------------------------
extern "C" void kernel_launch(void* const* d_in, const int* in_sizes, int n_in,
                              void* d_out, int out_size, void* d_ws, size_t ws_size,
                              hipStream_t stream) {
  (void)in_sizes; (void)n_in; (void)out_size; (void)ws_size;
  const float* inputs = (const float*)d_in[0];
  const float* pad    = (const float*)d_in[1];
  const float* Wx     = (const float*)d_in[2];
  const float* Wh     = (const float*)d_in[3];
  const float* bias   = (const float*)d_in[4];
  float* out = (float*)d_out;

  char* ws = (char*)d_ws;
  size_t off = 0;
  auto alloc = [&](size_t bytes) -> char* {
    char* p = ws + off;
    off += (bytes + 255) & ~(size_t)255;
    return p;
  };
  unsigned short* wxp  = (unsigned short*)alloc(6UL * 128 * 32 * 512 * 2);
  unsigned short* whp  = (unsigned short*)alloc(6UL * 1048576 * 2);
  unsigned short* xbfA = (unsigned short*)alloc((size_t)BB * TT * 1024 * 2);
  unsigned short* xbfB = (unsigned short*)alloc((size_t)BB * TT * 1024 * 2);
  float*          xgw  = (float*)alloc((size_t)BB * TT * (size_t)G4 * 4);
  unsigned short* hbuf = (unsigned short*)alloc(2UL * BB * HH * 2);
  int*            lens = (int*)alloc(BB * 4);
  unsigned int*   bar  = (unsigned int*)alloc(8 * 4);

  k_init<<<1, 64, 0, stream>>>(pad, lens, bar);
  {
    long tot = 6L * 128 * 32 * 512;
    k_pack_wx<<<(int)((tot + 255) / 256), 256, 0, stream>>>(Wx, wxp);
  }
  {
    long tot = 6L * 16 * 4 * 2 * 16 * 512;
    k_pack_wh<<<(int)((tot + 255) / 256), 256, 0, stream>>>(Wh, whp);
  }
  {
    long tot = (long)BB * TT * 1024;
    k_cvt<<<(int)((tot + 255) / 256), 256, 0, stream>>>(inputs, xbfA, tot);
  }

  unsigned short* xin  = xbfA;
  unsigned short* xnxt = xbfB;
  for (int l = 0; l < LL; ++l) {
    int lastl = (l == LL - 1);
    for (int d = 0; d < 2; ++d) {
      int ld = l * 2 + d;
      dim3 grid(256, 16, 1);
      k_xproj<<<grid, 256, 0, stream>>>(
          xin, wxp + (size_t)ld * 128 * 32 * 512, bias + (size_t)ld * G4, xgw);
      k_scan<<<NWG, 256, 0, stream>>>(
          xgw, whp + (size_t)ld * 1048576, hbuf + (size_t)d * BB * HH,
          xnxt, lastl ? out : (float*)nullptr, lens, bar + ld, d, lastl);
    }
    unsigned short* tmp = xin; xin = xnxt; xnxt = tmp;
  }
}
